// HyperbolicProjector_24043226923962
// MI455X (gfx1250) — matrix operand order repaired and checked by reading, not compiled, not run on hardware
//
#include <hip/hip_runtime.h>
#include <hip/hip_bf16.h>
#include <math.h>

// MI455X / gfx1250, wave32. Fused 1x1conv+BN+ReLU+proj+Poincare, bf16 WMMA.

typedef __attribute__((ext_vector_type(16))) __bf16 v16bf;
typedef __attribute__((ext_vector_type(8)))  float  v8f;

#define TOT_ROWS 8400

static __device__ __forceinline__ __bf16 f2bf(float f) { return (__bf16)f; }

union Frag16 { uint4 q[2]; v16bf v; };

static __device__ __forceinline__ v8f wmma_bf16(v16bf a, v16bf b, v8f c) {
  // (neg_a, A, neg_b, B, c_mod, C, reuse_a, reuse_b)
  return __builtin_amdgcn_wmma_f32_16x16x32_bf16(false, a, false, b, (short)0, c, false, false);
}

// ---------------- weight prep: fold BN into w1, convert to bf16 ----------------

__global__ void prep_w1_bn(const float* __restrict__ w1, const float* __restrict__ gamma,
                           const float* __restrict__ var, __bf16* __restrict__ w1b, int C) {
  int i = blockIdx.x * 256 + threadIdx.x;
  if (i >= C * C) return;
  int o = i / C;
  float inv = gamma[o] * rsqrtf(var[o] + 1e-5f);
  w1b[i] = f2bf(w1[i] * inv);
}

__global__ void prep_bias(const float* __restrict__ gamma, const float* __restrict__ beta,
                          const float* __restrict__ mean, const float* __restrict__ var,
                          float* __restrict__ bias, int C) {
  int o = blockIdx.x * 256 + threadIdx.x;
  if (o >= C) return;
  float inv = gamma[o] * rsqrtf(var[o] + 1e-5f);
  bias[o] = beta[o] - mean[o] * inv;
}

__global__ void prep_w2(const float* __restrict__ w2, __bf16* __restrict__ w2b, int n) {
  int i = blockIdx.x * 256 + threadIdx.x;
  if (i < n) w2b[i] = f2bf(w2[i]);
}

// ---------------- fused branch kernel ----------------
// 256 threads = 8 waves. Tile: 32 rows (M) x full C.
// GEMM1: h = relu(x @ w1b^T + bias)  -> LDS (bf16)
// GEMM2: z = h @ w2b^T (N=64)        -> LDS (f32)
// Epilogue: pre-expmap norms + Poincare expmap/project, coalesced store.

template<int C>
__global__ __launch_bounds__(256) void fused_branch(
    const float* __restrict__ x, const __bf16* __restrict__ w1b,
    const float* __restrict__ bias, const __bf16* __restrict__ w2b,
    float* __restrict__ outP, float* __restrict__ outN,
    int HW, int rowOff)
{
  constexpr int NT = C / 128;                 // n-tiles of 16 per wave (8 waves * NT*16 == C)
  __shared__ __align__(16) __bf16 xs[32][40]; // 32m x 32k chunk, +8 pad
  __shared__ __align__(16) __bf16 hs[32][C + 8];
  __shared__ float zs[32][66];                // 64 cols + pad; [64] holds expmap factor

  const int tid   = threadIdx.x;
  const int wave  = tid >> 5;
  const int lane  = tid & 31;
  const int lhalf = lane >> 4;                // 0: lanes 0-15, 1: lanes 16-31
  const int l15   = lane & 15;
  const int m0    = blockIdx.x * 32;

  const v8f vzero = {0.f, 0.f, 0.f, 0.f, 0.f, 0.f, 0.f, 0.f};

  v8f acc[2][NT];
  #pragma unroll
  for (int mt = 0; mt < 2; ++mt)
    #pragma unroll
    for (int nt = 0; nt < NT; ++nt)
      acc[mt][nt] = vzero;

  // cooperative x-chunk loader mapping: thread -> (k within chunk, group of 4 m)
  const int kk   = tid >> 3;                  // 0..31
  const int mloc = (tid & 7) * 4;             // 0,4,..,28
  const int mg   = m0 + mloc;
  const int bb   = mg / HW;                   // HW%4==0 -> float4 group stays in one batch
  const int sg   = mg % HW;
  const float* __restrict__ xrowbase = x + (size_t)bb * C * HW + sg;

  // ---- GEMM1 ----
  for (int k0 = 0; k0 < C; k0 += 32) {
    {
      const float4 v = *(const float4*)(xrowbase + (size_t)(k0 + kk) * HW);
      xs[mloc + 0][kk] = f2bf(v.x);
      xs[mloc + 1][kk] = f2bf(v.y);
      xs[mloc + 2][kk] = f2bf(v.z);
      xs[mloc + 3][kk] = f2bf(v.w);
    }
    __syncthreads();

    Frag16 a[2];
    #pragma unroll
    for (int mt = 0; mt < 2; ++mt) {
      const __bf16* ap = &xs[mt * 16 + l15][lhalf * 8];   // K {0..7}/{8..15}
      a[mt].q[0] = *(const uint4*)ap;
      a[mt].q[1] = *(const uint4*)(ap + 16);              // K {16..23}/{24..31}
    }
    #pragma unroll
    for (int nt = 0; nt < NT; ++nt) {
      const int o = (wave * NT + nt) * 16 + l15;          // B column = w1 row
      const __bf16* bp = w1b + (size_t)o * C + k0 + lhalf * 8; // K {0..7}/{8..15}, +16
      Frag16 bf;
      bf.q[0] = *(const uint4*)bp;
      bf.q[1] = *(const uint4*)(bp + 16);
      #pragma unroll
      for (int mt = 0; mt < 2; ++mt)
        acc[mt][nt] = wmma_bf16(a[mt].v, bf.v, acc[mt][nt]);
    }
    __syncthreads();
  }

  // GEMM1 epilogue: bias + relu -> hs (bf16). C/D layout: VGPR r -> M=r (lanes<16), 8+r else.
  #pragma unroll
  for (int nt = 0; nt < NT; ++nt) {
    const int o = (wave * NT + nt) * 16 + l15;
    const float bia = bias[o];
    #pragma unroll
    for (int mt = 0; mt < 2; ++mt) {
      const int mrow = mt * 16 + lhalf * 8;
      #pragma unroll
      for (int r = 0; r < 8; ++r) {
        float vv = acc[mt][nt][r] + bia;
        vv = fmaxf(vv, 0.f);
        hs[mrow + r][o] = f2bf(vv);
      }
    }
  }
  __syncthreads();

  // ---- GEMM2: one 16x16 tile of z per wave (2 m-tiles x 4 n-tiles) ----
  {
    const int mt2 = wave & 1;
    const int nt2 = wave >> 1;
    v8f acc2 = vzero;
    const __bf16* arow = &hs[mt2 * 16 + l15][lhalf * 8];
    const __bf16* brow = w2b + (size_t)(nt2 * 16 + l15) * C + lhalf * 8;
    for (int k0 = 0; k0 < C; k0 += 32) {
      Frag16 a2, b2;
      a2.q[0] = *(const uint4*)(arow + k0);
      a2.q[1] = *(const uint4*)(arow + k0 + 16);
      b2.q[0] = ((const uint4*)(brow + k0))[0];
      b2.q[1] = *(const uint4*)(brow + k0 + 16);
      acc2 = wmma_bf16(a2.v, b2.v, acc2);
    }
    const int mrow = mt2 * 16 + lhalf * 8;
    #pragma unroll
    for (int r = 0; r < 8; ++r)
      zs[mrow + r][nt2 * 16 + l15] = acc2[r];
  }
  __syncthreads();

  // ---- pre-expmap norms + Poincare factor (one lane per row) ----
  if (tid < 32) {
    const int m = m0 + tid;
    float ss = 0.f;
    #pragma unroll
    for (int jj = 0; jj < 64; ++jj) { const float t = zs[tid][jj]; ss += t * t; }
    const float n = sqrtf(ss);
    const int ob = m / HW, os = m % HW;
    outN[(size_t)ob * TOT_ROWS + rowOff + os] = n;

    const float scale = fminf(1.f, 2.f / fmaxf(n, 1e-15f));   // clip_r = 2, c = 1
    const float un    = fmaxf(n * scale, 1e-15f);
    const float th    = tanhf(un);
    float fac = (th / un) * scale;                            // expmap0 factor on original z
    const float gn = fmaxf(fabsf(th), 1e-15f);                // ||g|| == |tanh(un)|
    if (gn > 0.999f) fac *= 0.999f / gn;                      // project (never triggers: tanh<0.999)
    zs[tid][64] = fac;
  }
  __syncthreads();

  // ---- coalesced output of poincare embeddings ----
  #pragma unroll
  for (int i = 0; i < 8; ++i) {
    const int idx = tid + i * 256;            // 32*64 = 2048 = 8*256
    const int ml  = idx >> 6;
    const int jj  = idx & 63;
    const int m   = m0 + ml;
    const int ob  = m / HW, os = m % HW;
    outP[((size_t)ob * TOT_ROWS + rowOff + os) * 64 + jj] = zs[ml][64] * zs[ml][jj];
  }
}

// ---------------- launch ----------------

extern "C" void kernel_launch(void* const* d_in, const int* in_sizes, int n_in,
                              void* d_out, int out_size, void* d_ws, size_t ws_size,
                              hipStream_t stream) {
  (void)in_sizes; (void)n_in; (void)out_size; (void)ws_size;

  const float* p3   = (const float*)d_in[0];
  const float* p4   = (const float*)d_in[1];
  const float* p5   = (const float*)d_in[2];
  const float* w1_3 = (const float*)d_in[3];
  const float* w2_3 = (const float*)d_in[4];
  const float* g3   = (const float*)d_in[5];
  const float* b3   = (const float*)d_in[6];
  const float* m3   = (const float*)d_in[7];
  const float* v3   = (const float*)d_in[8];
  const float* w1_4 = (const float*)d_in[9];
  const float* w2_4 = (const float*)d_in[10];
  const float* g4   = (const float*)d_in[11];
  const float* b4   = (const float*)d_in[12];
  const float* m4   = (const float*)d_in[13];
  const float* v4   = (const float*)d_in[14];
  const float* w1_5 = (const float*)d_in[15];
  const float* w2_5 = (const float*)d_in[16];
  const float* g5   = (const float*)d_in[17];
  const float* b5   = (const float*)d_in[18];
  const float* m5   = (const float*)d_in[19];
  const float* v5   = (const float*)d_in[20];

  // workspace layout (bytes), all offsets 256-aligned; total ~2.9 MB
  char* ws = (char*)d_ws;
  __bf16* w1b3 = (__bf16*)(ws + 0);        // 384*384*2  = 294912
  __bf16* w1b4 = (__bf16*)(ws + 294912);   // 768*768*2  = 1179648
  __bf16* w1b5 = (__bf16*)(ws + 1474560);  // 768*768*2  = 1179648
  __bf16* w2b3 = (__bf16*)(ws + 2654208);  // 64*384*2   = 49152
  __bf16* w2b4 = (__bf16*)(ws + 2703360);  // 64*768*2   = 98304
  __bf16* w2b5 = (__bf16*)(ws + 2801664);  // 64*768*2   = 98304
  float*  bias3 = (float*)(ws + 2899968);  // 384*4
  float*  bias4 = (float*)(ws + 2901504);  // 768*4
  float*  bias5 = (float*)(ws + 2904576);  // 768*4

  prep_w1_bn<<<(384 * 384 + 255) / 256, 256, 0, stream>>>(w1_3, g3, v3, w1b3, 384);
  prep_w1_bn<<<(768 * 768 + 255) / 256, 256, 0, stream>>>(w1_4, g4, v4, w1b4, 768);
  prep_w1_bn<<<(768 * 768 + 255) / 256, 256, 0, stream>>>(w1_5, g5, v5, w1b5, 768);
  prep_w2<<<(64 * 384 + 255) / 256, 256, 0, stream>>>(w2_3, w2b3, 64 * 384);
  prep_w2<<<(64 * 768 + 255) / 256, 256, 0, stream>>>(w2_4, w2b4, 64 * 768);
  prep_w2<<<(64 * 768 + 255) / 256, 256, 0, stream>>>(w2_5, w2b5, 64 * 768);
  prep_bias<<<2, 256, 0, stream>>>(g3, b3, m3, v3, bias3, 384);
  prep_bias<<<3, 256, 0, stream>>>(g4, b4, m4, v4, bias4, 768);
  prep_bias<<<3, 256, 0, stream>>>(g5, b5, m5, v5, bias5, 768);

  float* outP = (float*)d_out;
  float* outN = outP + (size_t)8 * TOT_ROWS * 64;

  fused_branch<384><<<1600, 256, 0, stream>>>(p3, w1b3, bias3, w2b3, outP, outN, 6400, 0);
  fused_branch<768><<<400,  256, 0, stream>>>(p4, w1b4, bias4, w2b4, outP, outN, 1600, 6400);
  fused_branch<768><<<100,  256, 0, stream>>>(p5, w1b5, bias5, w2b5, outP, outN, 400,  8000);
}
